// SpeechDecoder_87668872446652
// MI455X (gfx1250) — compile-verified, matrix operand order checked by reading, not run on hardware
//
#include <hip/hip_runtime.h>
#include <hip/hip_bf16.h>

// ---------------------------------------------------------------------------
// Types for CDNA5 WMMA (wave32): v_wmma_f32_16x16x32_bf16
// ---------------------------------------------------------------------------
typedef __bf16 bf16_t;
typedef __attribute__((ext_vector_type(16))) __bf16 v16bf;
typedef __attribute__((ext_vector_type(8)))  float  v8f;

union BFrag {
    v16bf  v;
    bf16_t e[16];
};

__device__ __forceinline__ bf16_t f2bf(float f) {
    unsigned u = __float_as_uint(f);
    unsigned r = u + 0x7FFFu + ((u >> 16) & 1u);   // round-to-nearest-even
    unsigned short s = (unsigned short)(r >> 16);
    return __builtin_bit_cast(bf16_t, s);
}

__device__ __forceinline__ float sigm(float x)  { return 1.0f / (1.0f + __expf(-x)); }
__device__ __forceinline__ float tanh_f(float x){ return 1.0f - 2.0f / (__expf(2.0f * x) + 1.0f); }

#define BB   64      // batch
#define HH   512     // hidden
#define EE   512     // encoder dim
#define SS   512     // encoder seq
#define FF   80      // feature
#define TT   256     // target len

// ---------------------------------------------------------------------------
// Pack [Wih | Whh] -> bf16, K padded to Kp, gate-interleaved rows n' = 4*h+gate
// Also packs bias into the same row order.
// ---------------------------------------------------------------------------
__global__ void pack_weights_kernel(const float* __restrict__ Wih, int cin,
                                    const float* __restrict__ Whh,
                                    const float* __restrict__ bias,
                                    bf16_t* __restrict__ Wp,
                                    float* __restrict__ bpk,
                                    int Kp, int total)
{
    int idx = blockIdx.x * 256 + threadIdx.x;
    if (idx >= total) return;
    int np = idx / Kp;
    int k  = idx - np * Kp;
    int h  = np >> 2;
    int gate = np & 3;
    int src  = gate * HH + h;        // PyTorch gate order i,f,g,o in row blocks
    float v;
    if (k < cin)            v = Wih[(size_t)src * cin + k];
    else if (k < cin + HH)  v = Whh[(size_t)src * HH + (k - cin)];
    else                    v = 0.0f;
    Wp[idx] = f2bf(v);
    if (k == 0) bpk[np] = bias[src];
}

// ---------------------------------------------------------------------------
// One LSTM layer, one time step, fused GEMM (bf16 WMMA) + gate nonlinearity.
// Input activation = concat(xA[lenA], xB[lenB], hprev[512]), K padded to Kp.
// Output tile per workgroup: 64 (all of B) x 128 (N). grid = 2048/128 = 16.
// 8 waves; wave w owns N column tile w; computes 4 M tiles (whole batch).
// ---------------------------------------------------------------------------
__global__ __launch_bounds__(256)
void lstm_step_kernel(const float* __restrict__ xA, int lenA, int strideA,
                      const float* __restrict__ xB, int lenB,
                      const float* __restrict__ hprev,
                      const float* __restrict__ cprev,
                      const bf16_t* __restrict__ Wp,
                      const float* __restrict__ bp,
                      int Kp,
                      float* __restrict__ hout,
                      float* __restrict__ cout)
{
    __shared__ bf16_t ldsX[64 * 128];   // activation chunk, bf16  (16 KB)
    __shared__ float  ldsG[64 * 128];   // pre-activation gates    (32 KB)

    const int tid  = threadIdx.x;
    const int lane = tid & 31;
    const int wave = tid >> 5;          // 0..7 : N sub-tile
    const int half = lane >> 4;         // K-half selector (WMMA layout)
    const int mr   = lane & 15;         // row-in-tile / col-in-tile
    const int nb0  = blockIdx.x * 128;  // packed-row base for this WG
    const int nrow = nb0 + wave * 16 + mr;
    const int hBeg = lenA + lenB;       // start of h_prev segment
    const int hEnd = hBeg + HH;

    v8f acc[4] = {};

    for (int kc = 0; kc < Kp; kc += 128) {
        const int kcw = (Kp - kc) < 128 ? (Kp - kc) : 128;
        __syncthreads();   // previous chunk fully consumed
        // stage activation chunk into LDS as bf16
        for (int idx = tid; idx < 64 * kcw; idx += 256) {
            int bb = idx / kcw;
            int kl = idx - bb * kcw;
            int k  = kc + kl;
            float v;
            if (k < lenA)       v = xA[(size_t)bb * strideA + k];
            else if (k < hBeg)  v = xB[bb * EE + (k - lenA)];
            else if (k < hEnd)  v = hprev[bb * HH + (k - hBeg)];
            else                v = 0.0f;
            ldsX[bb * 128 + kl] = f2bf(v);
        }
        __syncthreads();
        const int ksteps = kcw >> 5;
        for (int ks = 0; ks < ksteps; ++ks) {
            // B fragment: per-lane two 8-element bf16 runs (k = 8h+s, 16+8h+s-8) from packed weights
            BFrag bfr;
            const bf16_t* wr = Wp + (size_t)nrow * Kp + (kc + ks * 32 + half * 8);
            #pragma unroll
            for (int j = 0; j < 8; ++j) bfr.e[j] = wr[j];
            #pragma unroll
            for (int j = 0; j < 8; ++j) bfr.e[8 + j] = wr[16 + j];
            #pragma unroll
            for (int mt = 0; mt < 4; ++mt) {
                BFrag afr;
                const bf16_t* xr = ldsX + (mt * 16 + mr) * 128 + ks * 32 + half * 8;
                #pragma unroll
                for (int j = 0; j < 8; ++j) afr.e[j] = xr[j];
                #pragma unroll
                for (int j = 0; j < 8; ++j) afr.e[8 + j] = xr[16 + j];
                acc[mt] = __builtin_amdgcn_wmma_f32_16x16x32_bf16(
                    false, afr.v, false, bfr.v, (short)0, acc[mt], false, false);
            }
        }
    }

    __syncthreads();
    // D layout: VGPR r, lanes 0-15 -> M=r, N=lane ; lanes 16-31 -> M=r+8, N=lane-16
    #pragma unroll
    for (int mt = 0; mt < 4; ++mt)
        #pragma unroll
        for (int r = 0; r < 8; ++r)
            ldsG[(mt * 16 + half * 8 + r) * 128 + wave * 16 + mr] = acc[mt][r];
    __syncthreads();

    // Fused cell update: this WG's 128 packed rows = hidden units [blk*32, blk*32+32)
    for (int idx = tid; idx < 64 * 32; idx += 256) {
        int bb = idx >> 5;
        int ul = idx & 31;
        int col = ul * 4;
        float gi = ldsG[bb * 128 + col + 0] + bp[nb0 + col + 0];
        float gf = ldsG[bb * 128 + col + 1] + bp[nb0 + col + 1];
        float gg = ldsG[bb * 128 + col + 2] + bp[nb0 + col + 2];
        float go = ldsG[bb * 128 + col + 3] + bp[nb0 + col + 3];
        int hg = blockIdx.x * 32 + ul;
        float cp = cprev[bb * HH + hg];
        float cn = sigm(gf) * cp + sigm(gi) * tanh_f(gg);
        float hn = sigm(go) * tanh_f(cn);
        cout[bb * HH + hg] = cn;
        hout[bb * HH + hg] = hn;
    }
}

// ---------------------------------------------------------------------------
// Attention + output projection; one workgroup per batch row.
// q = h2 @ Wq ; scores = q.enc/sqrt(E) + mask ; softmax ; ctx = w.enc ;
// pred = ctx @ Wfc + bfc  ->  d_out[b, t, :]
// ---------------------------------------------------------------------------
__global__ __launch_bounds__(256)
void attn_kernel(const float* __restrict__ h2,
                 const float* __restrict__ enc,
                 const int* __restrict__ lens,
                 const float* __restrict__ Wq,
                 const float* __restrict__ Wfc,
                 const float* __restrict__ bfc,
                 float* __restrict__ ctx_out,
                 float* __restrict__ out, int outRowStride)
{
    __shared__ float sh_h[HH];
    __shared__ float sh_q[EE];
    __shared__ float sh_sc[SS];
    __shared__ float sh_ctx[EE];
    __shared__ float red[8];

    const int b = blockIdx.x;
    const int tid = threadIdx.x;
    const int lane = tid & 31;
    const int wave = tid >> 5;
    const float* encb = enc + (size_t)b * SS * EE;

    for (int i = tid; i < HH; i += 256) sh_h[i] = h2[b * HH + i];
    __syncthreads();

    // q = h2 @ Wq  (Wq is [H,E] row-major; e contiguous across lanes)
    for (int e = tid; e < EE; e += 256) {
        float a = 0.0f;
        for (int h = 0; h < HH; ++h) a += sh_h[h] * Wq[h * EE + e];
        sh_q[e] = a;
    }
    __syncthreads();

    // scores: wave-per-row dot, lane reduction via shfl_xor (wave32)
    const int len = lens[b];
    for (int s = wave; s < SS; s += 8) {
        const float* er = encb + (size_t)s * EE;
        float p = 0.0f;
        for (int e = lane; e < EE; e += 32) p += sh_q[e] * er[e];
        #pragma unroll
        for (int o = 16; o; o >>= 1) p += __shfl_xor(p, o, 32);
        if (lane == 0)
            sh_sc[s] = p * 0.04419417382415922f + (s < len ? 0.0f : -1e9f);
    }
    __syncthreads();

    // softmax: max
    float m = -3.4e38f;
    for (int s = tid; s < SS; s += 256) m = fmaxf(m, sh_sc[s]);
    #pragma unroll
    for (int o = 16; o; o >>= 1) m = fmaxf(m, __shfl_xor(m, o, 32));
    if (lane == 0) red[wave] = m;
    __syncthreads();
    m = red[0];
    #pragma unroll
    for (int i = 1; i < 8; ++i) m = fmaxf(m, red[i]);

    // exp + sum
    float ssum = 0.0f;
    for (int s = tid; s < SS; s += 256) {
        float e = __expf(sh_sc[s] - m);
        sh_sc[s] = e;
        ssum += e;
    }
    #pragma unroll
    for (int o = 16; o; o >>= 1) ssum += __shfl_xor(ssum, o, 32);
    __syncthreads();                 // all reads of red[] for max done
    if (lane == 0) red[wave] = ssum;
    __syncthreads();
    float tot = 0.0f;
    #pragma unroll
    for (int i = 0; i < 8; ++i) tot += red[i];
    float inv = 1.0f / tot;
    for (int s = tid; s < SS; s += 256) sh_sc[s] *= inv;
    __syncthreads();

    // ctx = w . enc (coalesced over e)
    for (int e = tid; e < EE; e += 256) {
        float a = 0.0f;
        for (int s = 0; s < SS; ++s) a += sh_sc[s] * encb[(size_t)s * EE + e];
        sh_ctx[e] = a;
        ctx_out[b * EE + e] = a;
    }
    __syncthreads();

    // pred = ctx @ Wfc + bfc  (Wfc is [E,F] row-major; f contiguous)
    for (int f = tid; f < FF; f += 256) {
        float a = bfc[f];
        for (int e = 0; e < EE; ++e) a += sh_ctx[e] * Wfc[e * FF + f];
        out[(size_t)b * outRowStride + f] = a;
    }
}

// ---------------------------------------------------------------------------
// Host driver: pack weights, zero state, 255 sequential steps of 4 kernels.
// ---------------------------------------------------------------------------
extern "C" void kernel_launch(void* const* d_in, const int* in_sizes, int n_in,
                              void* d_out, int out_size, void* d_ws, size_t ws_size,
                              hipStream_t stream)
{
    (void)in_sizes; (void)n_in; (void)out_size; (void)ws_size;
    const float* tgt  = (const float*)d_in[0];
    const float* enc  = (const float*)d_in[1];
    const int*   lens = (const int*)d_in[2];
    const float* Wih0 = (const float*)d_in[3];
    const float* Whh0 = (const float*)d_in[4];
    const float* b0   = (const float*)d_in[5];
    const float* Wih1 = (const float*)d_in[6];
    const float* Whh1 = (const float*)d_in[7];
    const float* b1   = (const float*)d_in[8];
    const float* Wih2 = (const float*)d_in[9];
    const float* Whh2 = (const float*)d_in[10];
    const float* b2   = (const float*)d_in[11];
    const float* Wq   = (const float*)d_in[12];
    const float* Wfc  = (const float*)d_in[13];
    const float* bfc  = (const float*)d_in[14];

    const int Kp0 = 1120;   // 80 + 512 + 512 = 1104, padded to 35*32
    const int Kp12 = 1024;  // 512 + 512
    const size_t BH = (size_t)BB * HH;

    char* ws = (char*)d_ws;
    size_t off = 0;
    auto take = [&](size_t bytes) -> size_t {
        size_t o = off;
        off += (bytes + 255) & ~(size_t)255;
        return o;
    };
    bf16_t* Wp0 = (bf16_t*)(ws + take((size_t)2048 * Kp0 * 2));
    bf16_t* Wp1 = (bf16_t*)(ws + take((size_t)2048 * Kp12 * 2));
    bf16_t* Wp2 = (bf16_t*)(ws + take((size_t)2048 * Kp12 * 2));
    float*  bp0 = (float*)(ws + take(2048 * 4));
    float*  bp1 = (float*)(ws + take(2048 * 4));
    float*  bp2 = (float*)(ws + take(2048 * 4));
    float*  hbuf = (float*)(ws + take(3 * 2 * BH * 4));  // h[layer][parity][b][j]
    float*  cbuf = (float*)(ws + take(3 * 2 * BH * 4));
    float*  ctxb = (float*)(ws + take(2 * BH * 4));      // ctx[parity][b][e]

    hipMemsetAsync(hbuf, 0, 3 * 2 * BH * 4, stream);
    hipMemsetAsync(cbuf, 0, 3 * 2 * BH * 4, stream);
    hipMemsetAsync(ctxb, 0, 2 * BH * 4, stream);

    {
        int tot = 2048 * Kp0;
        pack_weights_kernel<<<(tot + 255) / 256, 256, 0, stream>>>(Wih0, FF + EE, Whh0, b0, Wp0, bp0, Kp0, tot);
    }
    {
        int tot = 2048 * Kp12;
        pack_weights_kernel<<<(tot + 255) / 256, 256, 0, stream>>>(Wih1, HH, Whh1, b1, Wp1, bp1, Kp12, tot);
        pack_weights_kernel<<<(tot + 255) / 256, 256, 0, stream>>>(Wih2, HH, Whh2, b2, Wp2, bp2, Kp12, tot);
    }

    float* outp = (float*)d_out;
    for (int t = 0; t < TT - 1; ++t) {
        int p0 = t & 1, p1 = p0 ^ 1;
        // layer 0: input = [x_t(80) | ctx(512) | h0_prev(512)]
        lstm_step_kernel<<<16, 256, 0, stream>>>(
            tgt + (size_t)t * FF, FF, TT * FF,
            ctxb + p0 * BH, EE,
            hbuf + (0 * 2 + p0) * BH, cbuf + (0 * 2 + p0) * BH,
            Wp0, bp0, Kp0,
            hbuf + (0 * 2 + p1) * BH, cbuf + (0 * 2 + p1) * BH);
        // layer 1: input = [h0_new(512) | h1_prev(512)]
        lstm_step_kernel<<<16, 256, 0, stream>>>(
            hbuf + (0 * 2 + p1) * BH, HH, HH,
            nullptr, 0,
            hbuf + (1 * 2 + p0) * BH, cbuf + (1 * 2 + p0) * BH,
            Wp1, bp1, Kp12,
            hbuf + (1 * 2 + p1) * BH, cbuf + (1 * 2 + p1) * BH);
        // layer 2
        lstm_step_kernel<<<16, 256, 0, stream>>>(
            hbuf + (1 * 2 + p1) * BH, HH, HH,
            nullptr, 0,
            hbuf + (2 * 2 + p0) * BH, cbuf + (2 * 2 + p0) * BH,
            Wp2, bp2, Kp12,
            hbuf + (2 * 2 + p1) * BH, cbuf + (2 * 2 + p1) * BH);
        // attention + projection -> d_out[:, t, :]
        attn_kernel<<<BB, 256, 0, stream>>>(
            hbuf + (2 * 2 + p1) * BH, enc, lens, Wq, Wfc, bfc,
            ctxb + p1 * BH, outp + (size_t)t * FF, (TT - 1) * FF);
    }
}
